// Seq2Seq_84232898609866
// MI455X (gfx1250) — compile-verified, matrix operand order checked by reading, not run on hardware
//
#include <hip/hip_runtime.h>
#include <math.h>

// ---------------------------------------------------------------------------
// Problem constants (match reference)
// ---------------------------------------------------------------------------
#define NN   8192      // nodes
#define EE   65536     // edges
#define HID  256
#define LAY  2
#define TT   3
#define FIN  3
#define NH   (NN*HID)

typedef unsigned short ushortT;

// ---------------------------------------------------------------------------
// WMMA types (CDNA5 gfx1250, wave32)
// ---------------------------------------------------------------------------
typedef __attribute__((ext_vector_type(16))) __bf16 v16bf;
typedef __attribute__((ext_vector_type(8)))  float  v8f;

union BFrag { v16bf v; uint4 q[2]; ushortT h[16]; };

__device__ __forceinline__ ushortT f2bf(float f) {
    unsigned u = __float_as_uint(f);
    u += 0x7fffu + ((u >> 16) & 1u);      // round-to-nearest-even
    return (ushortT)(u >> 16);
}

__device__ __forceinline__ float sigf(float x) { return 1.f / (1.f + __expf(-x)); }

__device__ __forceinline__ void atomicMaxF(float* addr, float val) {
    int* ai = (int*)addr;
    int old = __float_as_int(*addr);
    while (__int_as_float(old) < val) {
        int assumed = old;
        old = atomicCAS(ai, assumed, __float_as_int(val));
        if (old == assumed) break;
    }
}

// ---------------------------------------------------------------------------
// One-shot conversion kernels (outside the GEMM hot loop)
// ---------------------------------------------------------------------------

// Activations: f32 (rows x K, lda) -> bf16 row-major, row stride K32*32, zero pad
__global__ void cvt_act_kernel(const float* __restrict__ in, int lda, int K,
                               ushortT* __restrict__ out, int K32, int rows)
{
    int i = blockIdx.x * blockDim.x + threadIdx.x;
    int ldab = K32 * 32;
    if (i >= rows * ldab) return;
    int r = i / ldab, c = i % ldab;
    float v = (c < K) ? in[(size_t)r * lda + c] : 0.f;
    out[i] = f2bf(v);
}

// Weights: f32 K x 256 row-major -> fragment-packed bf16 [ntile][kc][lane][16]
// so each lane's v16bf B fragment is 32 contiguous bytes (2 x b128 loads).
__global__ void cvt_wt_kernel(const float* __restrict__ W, int K,
                              ushortT* __restrict__ out, int K32)
{
    int i = blockIdx.x * blockDim.x + threadIdx.x;
    int total = 16 * K32 * 32;
    if (i >= total) return;
    int lane  = i & 31;
    int kc    = (i >> 5) % K32;
    int ntile = i / (32 * K32);
    int g = lane >> 4, colt = lane & 15;
    int col = ntile * 16 + colt;
    ushortT tmp[16];
    #pragma unroll
    for (int j = 0; j < 8; j++) {
        int kb = ((j < 4) ? 2 * j : 16 + 2 * (j - 4)) + g * 8 + kc * 32;
        float v0 = (kb     < K) ? W[(size_t)kb       * 256 + col] : 0.f;
        float v1 = (kb + 1 < K) ? W[(size_t)(kb + 1) * 256 + col] : 0.f;
        tmp[2 * j]     = f2bf(v0);
        tmp[2 * j + 1] = f2bf(v1);
    }
    ushortT* o = out + ((size_t)i << 4);
    #pragma unroll
    for (int j = 0; j < 16; j++) o[j] = tmp[j];
}

// ---------------------------------------------------------------------------
// GEMM:  Y(N x 256) = A(N x K) @ W(K x 256) + bias, all-bf16 inputs, f32 acc.
// Block = 128 threads (4 waves) covering a 32-row x 256-col output tile.
// Each wave: 2 M-tiles x 4 N-tiles = 8 WMMA per K-chunk. No LDS, no branches:
// A fragments are two aligned b128 loads per row (pre-padded bf16), B
// fragments are two b128 loads from the fragment-packed weights.
// ---------------------------------------------------------------------------
__global__ __launch_bounds__(128)
void gemm_bf16_wmma(const ushortT* __restrict__ A,    // N x (K32*32) bf16
                    const ushortT* __restrict__ Wf,   // fragment-packed
                    const float* __restrict__ bias,
                    float* __restrict__ Y, int K32)
{
    const int lane  = threadIdx.x & 31;
    const int wave  = threadIdx.x >> 5;
    const int g     = lane >> 4;
    const int lrow  = lane & 15;
    const int mbase = blockIdx.x * 32;
    const int ldab  = K32 * 32;

    v8f acc[2][4] = {};

    // per-lane A row pointers (uint4 = 8 bf16); fragment = [kc*4] and [kc*4+2]
    const uint4* a0 = (const uint4*)(A + (size_t)(mbase + lrow)      * ldab + g * 8);
    const uint4* a1 = (const uint4*)(A + (size_t)(mbase + 16 + lrow) * ldab + g * 8);

    for (int kc = 0; kc < K32; kc++) {
        BFrag af0, af1;
        af0.q[0] = a0[kc * 4];     af0.q[1] = a0[kc * 4 + 2];
        af1.q[0] = a1[kc * 4];     af1.q[1] = a1[kc * 4 + 2];
        #pragma unroll
        for (int t = 0; t < 4; t++) {
            const uint4* bp = (const uint4*)(Wf +
                ((((size_t)(wave * 4 + t) * K32 + kc) * 32 + lane) << 4));
            BFrag bf;
            bf.q[0] = bp[0];
            bf.q[1] = bp[1];
            acc[0][t] = __builtin_amdgcn_wmma_f32_16x16x32_bf16(false, af0.v, false, bf.v,
                                                               (short)0, acc[0][t], false, false);
            acc[1][t] = __builtin_amdgcn_wmma_f32_16x16x32_bf16(false, af1.v, false, bf.v,
                                                               (short)0, acc[1][t], false, false);
        }
    }

    #pragma unroll
    for (int mt = 0; mt < 2; mt++) {
        #pragma unroll
        for (int t = 0; t < 4; t++) {
            int col = (wave * 4 + t) * 16 + lrow;
            float bcol = bias[col];
            #pragma unroll
            for (int r = 0; r < 8; r++) {
                int row = mbase + mt * 16 + g * 8 + r;
                Y[(size_t)row * 256 + col] = acc[mt][t][r] + bcol;
            }
        }
    }
}

// ---------------------------------------------------------------------------
// Segment-softmax / aggregation kernels (256-dim path, fp32)
// ---------------------------------------------------------------------------
__global__ void init_seg_kernel(float* m, float* ssum, int n) {
    int i = blockIdx.x * blockDim.x + threadIdx.x;
    if (i < n) { m[i] = -1e30f; ssum[i] = 0.f; }
}

__global__ void edge_logits_kernel(const float* __restrict__ q, const float* __restrict__ k,
                                   const float* __restrict__ We, const float* __restrict__ be,
                                   const float* __restrict__ ea,
                                   const int* __restrict__ src, const int* __restrict__ dst,
                                   float* __restrict__ elog, float* __restrict__ m,
                                   float scale, int E)
{
    int wave = threadIdx.x >> 5, lane = threadIdx.x & 31;
    int e = blockIdx.x * 8 + wave;
    if (e >= E) return;
    int s = src[e], d = dst[e];
    float ea0 = ea[2 * e], ea1 = ea[2 * e + 1];
    float acc = 0.f;
    #pragma unroll
    for (int j = 0; j < 8; j++) {
        int f = lane + 32 * j;
        float ke = k[(size_t)s * 256 + f] + ea0 * We[f] + ea1 * We[256 + f] + be[f];
        acc += q[(size_t)d * 256 + f] * ke;
    }
    #pragma unroll
    for (int o = 16; o > 0; o >>= 1) acc += __shfl_xor(acc, o);
    if (lane == 0) {
        float lg = acc * scale;
        elog[e] = lg;
        atomicMaxF(&m[d], lg);
    }
}

__global__ void edge_exp_kernel(float* __restrict__ elog, const float* __restrict__ m,
                                float* __restrict__ ssum, const int* __restrict__ dst, int E)
{
    int e = blockIdx.x * blockDim.x + threadIdx.x;
    if (e >= E) return;
    int d = dst[e];
    float ex = __expf(elog[e] - m[d]);
    elog[e] = ex;
    atomicAdd(&ssum[d], ex);
}

__global__ void edge_agg_kernel(const float* __restrict__ v,
                                const float* __restrict__ We, const float* __restrict__ be,
                                const float* __restrict__ ea,
                                const int* __restrict__ src, const int* __restrict__ dst,
                                const float* __restrict__ elog, const float* __restrict__ ssum,
                                float* __restrict__ out, int E)
{
    int wave = threadIdx.x >> 5, lane = threadIdx.x & 31;
    int e = blockIdx.x * 8 + wave;
    if (e >= E) return;
    int s = src[e], d = dst[e];
    float alpha = elog[e] / (ssum[d] + 1e-16f);
    float ea0 = ea[2 * e], ea1 = ea[2 * e + 1];
    #pragma unroll
    for (int j = 0; j < 8; j++) {
        int f = lane + 32 * j;
        float ef = ea0 * We[f] + ea1 * We[256 + f] + be[f];
        atomicAdd(&out[(size_t)d * 256 + f], (v[(size_t)s * 256 + f] + ef) * alpha);
    }
}

// ---------------------------------------------------------------------------
// Pointwise / normalization kernels
// ---------------------------------------------------------------------------
__global__ void add2_kernel(const float* a, const float* b, float* o, int n) {
    int i = blockIdx.x * blockDim.x + threadIdx.x;
    if (i < n) o[i] = a[i] + b[i];
}

__global__ void lstm_kernel(const float* gi, const float* gf, const float* gg, const float* go,
                            float* H, float* C, int n)
{
    int i = blockIdx.x * blockDim.x + threadIdx.x;
    if (i >= n) return;
    float iv = sigf(gi[i]), fv = sigf(gf[i]), gv = tanhf(gg[i]), ov = sigf(go[i]);
    float c = fv * C[i] + iv * gv;
    float h = ov * tanhf(c);
    C[i] = c;
    H[i] = h;
}

__global__ void ln_kernel(const float* __restrict__ in, int ldin,
                          const float* __restrict__ g, const float* __restrict__ b,
                          float* __restrict__ out, int ldout, int doRelu, int rows)
{
    int wave = threadIdx.x >> 5, lane = threadIdx.x & 31;
    int r = blockIdx.x * 8 + wave;
    if (r >= rows) return;
    float vals[8];
    float s = 0.f;
    #pragma unroll
    for (int j = 0; j < 8; j++) { vals[j] = in[(size_t)r * ldin + lane + 32 * j]; s += vals[j]; }
    #pragma unroll
    for (int o = 16; o > 0; o >>= 1) s += __shfl_xor(s, o);
    float mean = s * (1.f / 256.f);
    float t = 0.f;
    #pragma unroll
    for (int j = 0; j < 8; j++) { float d = vals[j] - mean; t += d * d; }
    #pragma unroll
    for (int o = 16; o > 0; o >>= 1) t += __shfl_xor(t, o);
    float inv = rsqrtf(t * (1.f / 256.f) + 1e-5f);
    #pragma unroll
    for (int j = 0; j < 8; j++) {
        int f = lane + 32 * j;
        float y = (vals[j] - mean) * inv * g[f] + b[f];
        if (doRelu) y = fmaxf(y, 0.f);
        out[(size_t)r * ldout + f] = y;
    }
}

__global__ void copy_skip_kernel(const float* skip, float* xcat, int n) {
    int i = blockIdx.x * blockDim.x + threadIdx.x;
    if (i < n) {
        xcat[(size_t)i * 258 + 256] = skip[2 * i];
        xcat[(size_t)i * 258 + 257] = skip[2 * i + 1];
    }
}

// ---------------------------------------------------------------------------
// fc2 path (dout = 1): scalar tconv
// ---------------------------------------------------------------------------
__global__ void fc2_gemv_kernel(const float* __restrict__ X,
                                const float* Wq, const float* Wk, const float* Wv, const float* Ws,
                                const float* bq, const float* bk, const float* bv, const float* bs,
                                float* q1, float* k1, float* v1, float* o1, int n)
{
    int wave = threadIdx.x >> 5, lane = threadIdx.x & 31;
    int r = blockIdx.x * 8 + wave;
    if (r >= n) return;
    float sq = 0.f, sk = 0.f, sv = 0.f, ss = 0.f;
    #pragma unroll
    for (int j = 0; j < 8; j++) {
        int f = lane + 32 * j;
        float xv = fmaxf(X[(size_t)r * 256 + f], 0.f);
        sq += xv * Wq[f]; sk += xv * Wk[f]; sv += xv * Wv[f]; ss += xv * Ws[f];
    }
    #pragma unroll
    for (int o = 16; o > 0; o >>= 1) {
        sq += __shfl_xor(sq, o); sk += __shfl_xor(sk, o);
        sv += __shfl_xor(sv, o); ss += __shfl_xor(ss, o);
    }
    if (lane == 0) {
        q1[r] = sq + bq[0]; k1[r] = sk + bk[0];
        v1[r] = sv + bv[0]; o1[r] = ss + bs[0];
    }
}

__global__ void edge_scalar_logit_kernel(const float* q1, const float* k1,
                                         const float* We, const float* be, const float* ea,
                                         const int* src, const int* dst,
                                         float* elog, float* m, int E)
{
    int e = blockIdx.x * blockDim.x + threadIdx.x;
    if (e >= E) return;
    int s = src[e], d = dst[e];
    float e1 = ea[2 * e] * We[0] + ea[2 * e + 1] * We[1] + be[0];
    float lg = q1[d] * (k1[s] + e1);
    elog[e] = lg;
    atomicMaxF(&m[d], lg);
}

__global__ void edge_scalar_agg_kernel(const float* v1,
                                       const float* We, const float* be, const float* ea,
                                       const int* src, const int* dst,
                                       const float* elog, const float* ssum,
                                       float* o1, int E)
{
    int e = blockIdx.x * blockDim.x + threadIdx.x;
    if (e >= E) return;
    int s = src[e], d = dst[e];
    float alpha = elog[e] / (ssum[d] + 1e-16f);
    float e1 = ea[2 * e] * We[0] + ea[2 * e + 1] * We[1] + be[0];
    atomicAdd(&o1[d], (v1[s] + e1) * alpha);
}

__global__ void sigmoid_store_kernel(const float* o1, float* y, int n) {
    int i = blockIdx.x * blockDim.x + threadIdx.x;
    if (i < n) y[i] = sigf(o1[i]);
}

__global__ void build_xt_kernel(const float* y, const float* x0, float* xt, int n) {
    int i = blockIdx.x * blockDim.x + threadIdx.x;
    if (i < n) {
        xt[(size_t)i * 3 + 0] = y[i];
        xt[(size_t)i * 3 + 1] = x0[(size_t)i * 3 + 1];
        xt[(size_t)i * 3 + 2] = x0[(size_t)i * 3 + 2];
    }
}

// ---------------------------------------------------------------------------
// Host-side orchestration
// ---------------------------------------------------------------------------
struct TC  { const float *We, *Wk, *Wq, *Ws, *Wv, *be, *bk, *bq, *bs, *bv; };
struct TCB { const ushortT *Wq, *Wk, *Wv, *Ws;
             const float *We, *be, *bq, *bk, *bv, *bs; int K32; };

// jax tree_flatten sorts dict keys: We,Wk,Wq,Ws,Wv,be,bk,bq,bs,bv
static TC tcAt(void* const* d, int b) {
    TC t;
    t.We = (const float*)d[b + 0]; t.Wk = (const float*)d[b + 1];
    t.Wq = (const float*)d[b + 2]; t.Ws = (const float*)d[b + 3];
    t.Wv = (const float*)d[b + 4]; t.be = (const float*)d[b + 5];
    t.bk = (const float*)d[b + 6]; t.bq = (const float*)d[b + 7];
    t.bs = (const float*)d[b + 8]; t.bv = (const float*)d[b + 9];
    return t;
}

static void tconvB(hipStream_t st, const TCB& p, const ushortT* act,
                   const int* src, const int* dst, const float* ea,
                   float* q, float* k, float* v, float* out,
                   float* elog, float* m, float* ssum)
{
    gemm_bf16_wmma<<<NN / 32, 128, 0, st>>>(act, p.Wq, p.bq, q,   p.K32);
    gemm_bf16_wmma<<<NN / 32, 128, 0, st>>>(act, p.Wk, p.bk, k,   p.K32);
    gemm_bf16_wmma<<<NN / 32, 128, 0, st>>>(act, p.Wv, p.bv, v,   p.K32);
    gemm_bf16_wmma<<<NN / 32, 128, 0, st>>>(act, p.Ws, p.bs, out, p.K32);
    init_seg_kernel<<<(NN + 255) / 256, 256, 0, st>>>(m, ssum, NN);
    edge_logits_kernel<<<(EE + 7) / 8, 256, 0, st>>>(q, k, p.We, p.be, ea, src, dst,
                                                     elog, m, 1.f / 16.f, EE);
    edge_exp_kernel<<<(EE + 255) / 256, 256, 0, st>>>(elog, m, ssum, dst, EE);
    edge_agg_kernel<<<(EE + 7) / 8, 256, 0, st>>>(v, p.We, p.be, ea, src, dst,
                                                  elog, ssum, out, EE);
}

extern "C" void kernel_launch(void* const* d_in, const int* in_sizes, int n_in,
                              void* d_out, int out_size, void* d_ws, size_t ws_size,
                              hipStream_t stream)
{
    (void)in_sizes; (void)n_in; (void)out_size; (void)ws_size;

    const float* x0   = (const float*)d_in[0];
    const int*   src  = (const int*)  d_in[1];
    const int*   dst  = (const int*)  d_in[2];
    const float* ea   = (const float*)d_in[3];
    const float* skip = (const float*)d_in[4];
    const float* H0   = (const float*)d_in[5];
    const float* C0   = (const float*)d_in[6];

    TC fc1 = tcAt(d_in, 7);
    TC fc2 = tcAt(d_in, 17);
    const float* nc_b = (const float*)d_in[27];
    const float* nc_g = (const float*)d_in[28];
    const float* nh_b = (const float*)d_in[29];
    const float* nh_g = (const float*)d_in[30];
    const float* no_b = (const float*)d_in[31];
    const float* no_g = (const float*)d_in[32];
    TC xg_[LAY][4], hg_[LAY][4];
    for (int l = 0; l < LAY; l++) {
        int base = 33 + l * 80;               // hf hg hi ho xf xg xi xo
        hg_[l][0] = tcAt(d_in, base + 20);
        hg_[l][1] = tcAt(d_in, base +  0);
        hg_[l][2] = tcAt(d_in, base + 10);
        hg_[l][3] = tcAt(d_in, base + 30);
        xg_[l][0] = tcAt(d_in, base + 60);
        xg_[l][1] = tcAt(d_in, base + 40);
        xg_[l][2] = tcAt(d_in, base + 50);
        xg_[l][3] = tcAt(d_in, base + 70);
    }

    // -------- workspace carve-up (256B aligned blocks) --------
    char* p = (char*)d_ws;
    auto allocB = [&](size_t bytes) { char* r = p; p += (bytes + 255) & ~(size_t)255; return r; };
    auto fAlloc = [&](size_t nf) { return (float*)allocB(nf * sizeof(float)); };
    auto hAlloc = [&](size_t nh) { return (ushortT*)allocB(nh * sizeof(ushortT)); };

    float* q    = fAlloc(NH);
    float* k    = fAlloc(NH);
    float* v    = fAlloc(NH);
    float* tmpA = fAlloc(NH);
    float* tmpB = fAlloc(NH);
    float* gate[4];
    for (int g = 0; g < 4; g++) gate[g] = fAlloc(NH);
    float* Hs   = fAlloc((size_t)LAY * NH);
    float* Cs   = fAlloc((size_t)LAY * NH);
    float* xcat = fAlloc((size_t)NN * 258);
    float* elog = fAlloc(EE);
    float* m    = fAlloc(NN);
    float* ssum = fAlloc(NN);
    float* q1   = fAlloc(NN);
    float* k1   = fAlloc(NN);
    float* v1   = fAlloc(NN);
    float* o1   = fAlloc(NN);
    float* xt   = fAlloc((size_t)NN * 3);
    ushortT* actX = hAlloc((size_t)NN * 288);   // up to K32=9
    ushortT* actH = hAlloc((size_t)NN * 256);

    // -------- convert all WMMA weights to fragment-packed bf16 --------
    auto cvtW = [&](const float* W, int K, int K32) -> const ushortT* {
        ushortT* dstw = hAlloc((size_t)16 * K32 * 32 * 16);
        int tot = 16 * K32 * 32;
        cvt_wt_kernel<<<(tot + 255) / 256, 256, 0, stream>>>(W, K, dstw, K32);
        return dstw;
    };
    auto mkTCB = [&](const TC& t, int K) -> TCB {
        TCB b; b.K32 = (K + 31) / 32;
        b.Wq = cvtW(t.Wq, K, b.K32); b.Wk = cvtW(t.Wk, K, b.K32);
        b.Wv = cvtW(t.Wv, K, b.K32); b.Ws = cvtW(t.Ws, K, b.K32);
        b.We = t.We; b.be = t.be; b.bq = t.bq; b.bk = t.bk; b.bv = t.bv; b.bs = t.bs;
        return b;
    };

    TCB fc1B = mkTCB(fc1, 258);
    TCB xB[LAY][4], hB[LAY][4];
    for (int l = 0; l < LAY; l++)
        for (int g = 0; g < 4; g++) {
            xB[l][g] = mkTCB(xg_[l][g], (l == 0) ? FIN : 256);
            hB[l][g] = mkTCB(hg_[l][g], 256);
        }

    // -------- state copies (do not mutate inputs) --------
    hipMemcpyAsync(Hs, H0, (size_t)LAY * NH * sizeof(float), hipMemcpyDeviceToDevice, stream);
    hipMemcpyAsync(Cs, C0, (size_t)LAY * NH * sizeof(float), hipMemcpyDeviceToDevice, stream);

    float* out = (float*)d_out;   // (T, N, 1)

    for (int t = 0; t < TT; t++) {
        const float* inp = (t == 0) ? x0 : xt;
        int lin = 3, Kin = FIN;

        for (int l = 0; l < LAY; l++) {
            float* Hl = Hs + (size_t)l * NH;
            float* Cl = Cs + (size_t)l * NH;
            int KinP = 32 * ((Kin + 31) / 32);

            // convert activations once per (layer, path)
            {
                int tot = NN * KinP;
                cvt_act_kernel<<<(tot + 255) / 256, 256, 0, stream>>>(inp, lin, Kin, actX, KinP / 32, NN);
            }
            cvt_act_kernel<<<(NH + 255) / 256, 256, 0, stream>>>(Hl, 256, 256, actH, 8, NN);

            for (int g = 0; g < 4; g++) {
                tconvB(stream, xB[l][g], actX, src, dst, ea, q, k, v, tmpA, elog, m, ssum);
                tconvB(stream, hB[l][g], actH, src, dst, ea, q, k, v, tmpB, elog, m, ssum);
                add2_kernel<<<(NH + 255) / 256, 256, 0, stream>>>(tmpA, tmpB, gate[g], NH);
            }
            lstm_kernel<<<(NH + 255) / 256, 256, 0, stream>>>(gate[0], gate[1], gate[2], gate[3],
                                                              Hl, Cl, NH);
            if (l == LAY - 1)   // y = relu(ln_o(h_raw)) before in-place ln_h
                ln_kernel<<<(NN + 7) / 8, 256, 0, stream>>>(Hl, 256, no_g, no_b, xcat, 258, 1, NN);
            ln_kernel<<<(NN + 7) / 8, 256, 0, stream>>>(Hl, 256, nh_g, nh_b, Hl, 256, 0, NN);
            ln_kernel<<<(NN + 7) / 8, 256, 0, stream>>>(Cl, 256, nc_g, nc_b, Cl, 256, 0, NN);
            inp = Hl; lin = 256; Kin = 256;
        }

        copy_skip_kernel<<<(NN + 255) / 256, 256, 0, stream>>>(skip, xcat, NN);

        // fc1 tconv on (N x 258)
        {
            int tot = NN * 288;
            cvt_act_kernel<<<(tot + 255) / 256, 256, 0, stream>>>(xcat, 258, 258, actX, 9, NN);
        }
        tconvB(stream, fc1B, actX, src, dst, ea, q, k, v, tmpA, elog, m, ssum);

        // fc2 scalar tconv (ReLU applied on load)
        fc2_gemv_kernel<<<(NN + 7) / 8, 256, 0, stream>>>(tmpA,
            fc2.Wq, fc2.Wk, fc2.Wv, fc2.Ws, fc2.bq, fc2.bk, fc2.bv, fc2.bs,
            q1, k1, v1, o1, NN);
        init_seg_kernel<<<(NN + 255) / 256, 256, 0, stream>>>(m, ssum, NN);
        edge_scalar_logit_kernel<<<(EE + 255) / 256, 256, 0, stream>>>(q1, k1, fc2.We, fc2.be,
                                                                       ea, src, dst, elog, m, EE);
        edge_exp_kernel<<<(EE + 255) / 256, 256, 0, stream>>>(elog, m, ssum, dst, EE);
        edge_scalar_agg_kernel<<<(EE + 255) / 256, 256, 0, stream>>>(v1, fc2.We, fc2.be,
                                                                     ea, src, dst, elog, ssum, o1, EE);
        sigmoid_store_kernel<<<(NN + 255) / 256, 256, 0, stream>>>(o1, out + (size_t)t * NN, NN);
        if (t < TT - 1)
            build_xt_kernel<<<(NN + 255) / 256, 256, 0, stream>>>(out + (size_t)t * NN, x0, xt, NN);
    }
}
